// VectorQuantizer_37993280701197
// MI455X (gfx1250) — matrix operand order repaired and checked by reading, not compiled, not run on hardware
//
#include <hip/hip_runtime.h>

typedef __attribute__((ext_vector_type(16))) __bf16 v16bf;
typedef __attribute__((ext_vector_type(8)))  __bf16 v8bf;
typedef __attribute__((ext_vector_type(8)))  float  v8f;
typedef int v4i_t __attribute__((vector_size(16)));   // matches async builtin param

#define DIM         64      // embedding dim
#define NUM_E       1024    // codebook entries
#define CP          256     // codes staged in LDS per pass
#define NPASS       (NUM_E / CP)
#define TOK_PER_WG  256     // 8 waves * 32 tokens
#define THREADS     256
#define CB_STRIDE   72      // bf16 elems per LDS/ws codebook row (64 + 8 pad -> 144B)

// ---------------------------------------------------------------------------
// CDNA5 async global->LDS copies (ASYNCcnt-tracked), with sync fallback.
// ---------------------------------------------------------------------------
#if defined(__AMDGCN__) && __has_builtin(__builtin_amdgcn_global_load_async_to_lds_b128)
#define ASYNC_LDS 1
#else
#define ASYNC_LDS 0
#endif

__device__ __forceinline__ void cp16_g2l(const void* g, void* l) {
#if ASYNC_LDS
    __builtin_amdgcn_global_load_async_to_lds_b128(
        (__attribute__((address_space(1))) v4i_t*)(void*)g,
        (__attribute__((address_space(3))) v4i_t*)l, 0, 0);
#else
    *(float4*)l = *(const float4*)g;
#endif
}

__device__ __forceinline__ void wait_async_copies() {
#if ASYNC_LDS
# if __has_builtin(__builtin_amdgcn_s_wait_asynccnt)
    __builtin_amdgcn_s_wait_asynccnt(0);
# else
    asm volatile("s_wait_asynccnt 0" ::: "memory");
# endif
#endif
}

// ---------------------------------------------------------------------------
// Prep kernel (runs once): fp32 codebook -> padded bf16 rows + 0.5*||e||^2
// ---------------------------------------------------------------------------
__global__ __launch_bounds__(256)
void vq_prep_kernel(const float* __restrict__ codebook,
                    __bf16* __restrict__ cbBf,     // [NUM_E * CB_STRIDE]
                    float* __restrict__ cnG)       // [NUM_E]
{
    const int c = blockIdx.x * 256 + threadIdx.x;  // 0..NUM_E-1
    const float* crow = codebook + (size_t)c * DIM;
    __bf16* dst = cbBf + (size_t)c * CB_STRIDE;
    float ss = 0.f;
    #pragma unroll
    for (int d = 0; d < DIM; d += 4) {
        const float4 u = *(const float4*)(crow + d);
        ss += u.x*u.x + u.y*u.y + u.z*u.z + u.w*u.w;
        dst[d+0] = (__bf16)u.x; dst[d+1] = (__bf16)u.y;
        dst[d+2] = (__bf16)u.z; dst[d+3] = (__bf16)u.w;
    }
    #pragma unroll
    for (int d = DIM; d < CB_STRIDE; ++d) dst[d] = (__bf16)0.f;  // pad
    cnG[c] = 0.5f * ss;
}

// ---------------------------------------------------------------------------
// Stage one 256-code pass into an LDS buffer (async): one padded row/thread.
// ---------------------------------------------------------------------------
__device__ __forceinline__ void stage_pass(const __bf16* __restrict__ cbBf,
                                           const float* __restrict__ cnG,
                                           __bf16* ldsCb, float* ldsCn,
                                           int p, int t)
{
    const __bf16* src = cbBf + (size_t)(p * CP + t) * CB_STRIDE;
    __bf16* dst = ldsCb + t * CB_STRIDE;
    #pragma unroll
    for (int i = 0; i < 9; ++i) cp16_g2l(src + i * 8, dst + i * 8);
    ldsCn[t] = cnG[p * CP + t];
}

// ---------------------------------------------------------------------------
// Build one 16-token A tile pair (K=0..31 and K=32..63) with hi/lo bf16 split.
// WMMA 16-bit A layout: lane M = l16; elem e<8 -> K = kb+8*hf+e,
// e>=8 -> K = kb+16+8*hf+(e-8).
// ---------------------------------------------------------------------------
__device__ __forceinline__ void build_A(const float* __restrict__ zr, int hf,
                                        v16bf& Ah0, v16bf& Al0,
                                        v16bf& Ah1, v16bf& Al1)
{
    float tmp0[16], tmp1[16];
    #pragma unroll
    for (int g = 0; g < 2; ++g) {   // g=0 -> K base 8*hf ; g=1 -> K base 16+8*hf
        const float4 u0 = *(const float4*)(zr + g*16 + 8*hf);
        const float4 u1 = *(const float4*)(zr + g*16 + 8*hf + 4);
        tmp0[g*8+0]=u0.x; tmp0[g*8+1]=u0.y; tmp0[g*8+2]=u0.z; tmp0[g*8+3]=u0.w;
        tmp0[g*8+4]=u1.x; tmp0[g*8+5]=u1.y; tmp0[g*8+6]=u1.z; tmp0[g*8+7]=u1.w;
        const float4 w0 = *(const float4*)(zr + 32 + g*16 + 8*hf);
        const float4 w1 = *(const float4*)(zr + 32 + g*16 + 8*hf + 4);
        tmp1[g*8+0]=w0.x; tmp1[g*8+1]=w0.y; tmp1[g*8+2]=w0.z; tmp1[g*8+3]=w0.w;
        tmp1[g*8+4]=w1.x; tmp1[g*8+5]=w1.y; tmp1[g*8+6]=w1.z; tmp1[g*8+7]=w1.w;
    }
    #pragma unroll
    for (int e = 0; e < 16; ++e) {
        __bf16 h0 = (__bf16)tmp0[e];
        Ah0[e] = h0;  Al0[e] = (__bf16)(tmp0[e] - (float)h0);
        __bf16 h1 = (__bf16)tmp1[e];
        Ah1[e] = h1;  Al1[e] = (__bf16)(tmp1[e] - (float)h1);
    }
}

// ---------------------------------------------------------------------------
// Main kernel: per 256-token workgroup compute argmax_k (z.e_k - 0.5||e_k||^2).
// 32 tokens per wave (two A tile-sets sharing each B tile -> half the LDS
// traffic per WMMA), double-buffered async staging, fused gather + loss.
// ---------------------------------------------------------------------------
__global__ __launch_bounds__(THREADS)
void vq_main_kernel(const float* __restrict__ z,
                    const float* __restrict__ codebook,   // fp32, for gather
                    const __bf16* __restrict__ cbBf,      // padded bf16 rows
                    const float* __restrict__ cnG,        // 0.5*||e||^2
                    float* __restrict__ zq_out,           // [N*DIM]
                    float* __restrict__ partial)          // [gridDim.x]
{
    __shared__ __bf16 cbL[2][CP * CB_STRIDE];  // 2 x 36 KB double buffer
    __shared__ float  cnL[2][CP];
    __shared__ int    sIdx[TOK_PER_WG];
    __shared__ float  sLoss[THREADS];

    const int t    = threadIdx.x;
    const int lane = t & 31;
    const int wave = t >> 5;
    const int hf   = lane >> 4;   // 16-lane half
    const int l16  = lane & 15;
    const int wgTokBase = blockIdx.x * TOK_PER_WG;
    const int tokBase   = wgTokBase + wave * 32;   // this wave's 32 tokens

    // Kick off pass 0 staging first so A-tile construction hides its latency.
    stage_pass(cbBf, cnG, cbL[0], cnL[0], 0, t);

    // Two 16-token A tile-sets (hi/lo split, K=0..31 and K=32..63 each).
    v16bf A0h, A0l, A1h, A1l;   // tokens tokBase + 0..15
    v16bf A2h, A2l, A3h, A3l;   // tokens tokBase + 16..31
    build_A(z + (size_t)(tokBase      + l16) * DIM, hf, A0h, A0l, A1h, A1l);
    build_A(z + (size_t)(tokBase + 16 + l16) * DIM, hf, A2h, A2l, A3h, A3l);

    wait_async_copies();
    __syncthreads();

    // Running best per (tile, acc vgpr v, lane): token M = v + 8*hf (+16*tile),
    // candidate codes restricted to residue class (code % 16 == l16).
    float bestS[2][8];
    int   bestI[2][8];
    #pragma unroll
    for (int v = 0; v < 8; ++v) {
        bestS[0][v] = -3.0e38f; bestI[0][v] = 0;
        bestS[1][v] = -3.0e38f; bestI[1][v] = 0;
    }

    for (int p = 0; p < NPASS; ++p) {
        const int buf = p & 1;
        // Issue next pass's async staging; overlaps with this pass's WMMAs.
        if (p + 1 < NPASS)
            stage_pass(cbBf, cnG, cbL[buf ^ 1], cnL[buf ^ 1], p + 1, t);

        for (int nb = 0; nb < CP; nb += 16) {
            const int r = nb + l16;                 // code row in LDS (N = l16)
            const __bf16* brow = &cbL[buf][r * CB_STRIDE];
            // B layout: lane elem e<8 -> K = kb+8*hf+e, e>=8 -> K = kb+16+8*hf+(e-8)
            v8bf b0a = *(const v8bf*)(brow + 8*hf);
            v8bf b0b = *(const v8bf*)(brow + 16 + 8*hf);
            v8bf b1a = *(const v8bf*)(brow + 32 + 8*hf);
            v8bf b1b = *(const v8bf*)(brow + 32 + 16 + 8*hf);
            v16bf B0 = __builtin_shufflevector(b0a, b0b, 0,1,2,3,4,5,6,7,8,9,10,11,12,13,14,15);
            v16bf B1 = __builtin_shufflevector(b1a, b1b, 0,1,2,3,4,5,6,7,8,9,10,11,12,13,14,15);

            // Two interleaved accumulator chains share B0/B1 (amortize LDS).
            v8f accA = {}, accB = {};   // SRC2 = inline 0 on first WMMA
            accA = __builtin_amdgcn_wmma_f32_16x16x32_bf16(false, A0h, false, B0, (short)0, accA, false, false);
            accB = __builtin_amdgcn_wmma_f32_16x16x32_bf16(false, A2h, false, B0, (short)0, accB, false, false);
            accA = __builtin_amdgcn_wmma_f32_16x16x32_bf16(false, A0l, false, B0, (short)0, accA, false, false);
            accB = __builtin_amdgcn_wmma_f32_16x16x32_bf16(false, A2l, false, B0, (short)0, accB, false, false);
            accA = __builtin_amdgcn_wmma_f32_16x16x32_bf16(false, A1h, false, B1, (short)0, accA, false, false);
            accB = __builtin_amdgcn_wmma_f32_16x16x32_bf16(false, A3h, false, B1, (short)0, accB, false, false);
            accA = __builtin_amdgcn_wmma_f32_16x16x32_bf16(false, A1l, false, B1, (short)0, accA, false, false);
            accB = __builtin_amdgcn_wmma_f32_16x16x32_bf16(false, A3l, false, B1, (short)0, accB, false, false);

            const float cn   = cnL[buf][r];         // 0.5*||e||^2, post-WMMA
            const int   cand = p * CP + r;
            #pragma unroll
            for (int v = 0; v < 8; ++v) {
                const float sA = accA[v] - cn;
                if (sA > bestS[0][v]) { bestS[0][v] = sA; bestI[0][v] = cand; }
                const float sB = accB[v] - cn;
                if (sB > bestS[1][v]) { bestS[1][v] = sB; bestI[1][v] = cand; }
            }
        }

        wait_async_copies();   // next buffer staged
        __syncthreads();       // all waves done with cbL[buf] before restage
    }

    // ---- Cross-lane argmax per token: butterfly over the 16-lane half ----
    #pragma unroll
    for (int tile = 0; tile < 2; ++tile) {
        #pragma unroll
        for (int v = 0; v < 8; ++v) {
            float s = bestS[tile][v];
            int   i = bestI[tile][v];
            #pragma unroll
            for (int off = 1; off < 16; off <<= 1) {
                const float so = __shfl_xor(s, off, 32);
                const int   io = __shfl_xor(i, off, 32);
                if (so > s || (so == s && io < i)) { s = so; i = io; }
            }
            if (l16 == 0)   // token m = tile*16 + v + 8*hf
                sIdx[wave * 32 + tile * 16 + hf * 8 + v] = i;
        }
    }
    __syncthreads();

    // ---- Gather z_q (fp32 codebook) + fused (zq - z)^2 partial sum ----
    {
        const int gtok = wgTokBase + t;     // one token per thread
        const int idx  = sIdx[t];
        const float* crow = codebook + (size_t)idx  * DIM;
        const float* zr   = z        + (size_t)gtok * DIM;
        float*       orow = zq_out   + (size_t)gtok * DIM;
        float ls = 0.f;
        #pragma unroll
        for (int d = 0; d < DIM; d += 4) {
            const float4 q  = *(const float4*)(crow + d);
            const float4 zz = *(const float4*)(zr + d);
            const float dx = q.x - zz.x, dy = q.y - zz.y;
            const float dz = q.z - zz.z, dw = q.w - zz.w;
            ls += dx*dx + dy*dy + dz*dz + dw*dw;
            *(float4*)(orow + d) = q;
        }
        sLoss[t] = ls;
    }
    __syncthreads();
    for (int r = THREADS / 2; r > 0; r >>= 1) {
        if (t < r) sLoss[t] += sLoss[t + r];
        __syncthreads();
    }
    if (t == 0) partial[blockIdx.x] = sLoss[0];
}

// ---------------------------------------------------------------------------
// Final deterministic reduction: loss = (1 + BETA) * mean((zq - z)^2)
// ---------------------------------------------------------------------------
__global__ __launch_bounds__(256)
void vq_finalize_kernel(const float* __restrict__ partial, int nPart,
                        float* __restrict__ lossOut, float scale)
{
    __shared__ float s[256];
    float acc = 0.f;
    for (int i = threadIdx.x; i < nPart; i += 256) acc += partial[i];
    s[threadIdx.x] = acc;
    __syncthreads();
    for (int r = 128; r > 0; r >>= 1) {
        if (threadIdx.x < r) s[threadIdx.x] += s[threadIdx.x + r];
        __syncthreads();
    }
    if (threadIdx.x == 0) *lossOut = s[0] * scale;
}

extern "C" void kernel_launch(void* const* d_in, const int* in_sizes, int n_in,
                              void* d_out, int out_size, void* d_ws, size_t ws_size,
                              hipStream_t stream) {
    const float* z        = (const float*)d_in[0];   // [N, 64] fp32
    const float* codebook = (const float*)d_in[1];   // [1024, 64] fp32
    float* out = (float*)d_out;                      // [N*64] z_q ++ [1] loss

    // Workspace layout: padded bf16 codebook | cnorm | per-block partials
    char* ws = (char*)d_ws;
    __bf16* cbBf   = (__bf16*)ws;                                   // 147456 B
    float*  cnG    = (float*)(ws + (size_t)NUM_E * CB_STRIDE * 2);  //   4096 B
    float*  partial= (float*)(ws + (size_t)NUM_E * CB_STRIDE * 2 + NUM_E * 4);

    const int nTok = in_sizes[0] / DIM;              // 131072
    const int grid = nTok / TOK_PER_WG;              // 512 workgroups

    vq_prep_kernel<<<NUM_E / 256, 256, 0, stream>>>(codebook, cbBf, cnG);
    vq_main_kernel<<<grid, THREADS, 0, stream>>>(z, codebook, cbBf, cnG, out, partial);
    vq_finalize_kernel<<<1, 256, 0, stream>>>(
        partial, grid, out + (size_t)nTok * DIM, 1.25f / (float)((size_t)nTok * DIM));
}
